// MultiHeadAttention_68822555951437
// MI455X (gfx1250) — matrix operand order repaired and checked by reading, not compiled, not run on hardware
//
#include <hip/hip_runtime.h>

// ---------------------------------------------------------------------------
// MultiHeadAttention for MI455X (gfx1250): bf16 WMMA everywhere, fp32 softmax,
// TDM (tensor_load_to_lds) staging for Q/K tiles in the flash-attention stage.
// B=4, S=2048, D=1024, H=16, DK=64.
// ---------------------------------------------------------------------------

typedef __bf16 bf16;
typedef __attribute__((ext_vector_type(16))) __bf16        v16bf;
typedef __attribute__((ext_vector_type(8)))  __bf16        v8bf;
typedef __attribute__((ext_vector_type(8)))  float         v8f;
typedef __attribute__((ext_vector_type(4)))  unsigned int  u32x4;
typedef __attribute__((ext_vector_type(8)))  int           i32x8;
typedef __attribute__((ext_vector_type(4)))  int           i32x4;

#define B_  4
#define S_  2048
#define D_  1024
#define H_  16
#define DK_ 64

#if defined(__gfx1250__) && __has_builtin(__builtin_amdgcn_tensor_load_to_lds) && \
    __has_builtin(__builtin_amdgcn_s_wait_tensorcnt)
#define HAVE_TDM 1
#else
#define HAVE_TDM 0
#endif

__device__ __forceinline__ v8f wmma_bf16(v16bf a, v16bf b, v8f c) {
    // emits v_wmma_f32_16x16x32_bf16
    return __builtin_amdgcn_wmma_f32_16x16x32_bf16(
        /*neg_a=*/false, a, /*neg_b=*/false, b,
        /*c_mod=*/(short)0, c, /*reuse_a=*/false, /*reuse_b=*/false);
}

// Build a 16-element fragment from two contiguous 8-element (16B) LDS runs.
__device__ __forceinline__ v16bf frag16(const bf16* p0, const bf16* p1) {
    v8bf lo = *(const v8bf*)p0;
    v8bf hi = *(const v8bf*)p1;
    v16bf r;
#pragma unroll
    for (int i = 0; i < 8; ++i) { r[i] = lo[i]; r[8 + i] = hi[i]; }
    return r;
}

#if HAVE_TDM
// ---------------------------------------------------------------------------
// TDM: DMA a 64x64 bf16 tile from a row-major [2048 x 64] tensor into LDS,
// with LDS row padding 64 -> 72 elements (pad 4 DWORDs after every 32 DWORDs),
// matching the __shared__ [64][72] layout used by the WMMA fragment gathers.
// D# layout per CDNA5 ISA (groups 0/1; 2D tile).
// ---------------------------------------------------------------------------
__device__ __forceinline__ void tdm_load_tile64(const bf16* gsrc, unsigned lds_addr) {
    unsigned long long ga = (unsigned long long)gsrc;
    u32x4 g0;
    g0[0] = 1u;                                       // count=1 (valid), no gather
    g0[1] = lds_addr;                                 // lds_addr (byte)
    g0[2] = (unsigned)(ga & 0xffffffffu);             // global_addr[31:0]
    g0[3] = (unsigned)((ga >> 32) & 0x01ffffffu)      // global_addr[56:32]
          | (2u << 30);                               // type = 2 ("image")
    i32x8 g1;
    g1[0] = (1 << 16)                                 // data_size = 2 bytes
          | (1 << 20)                                 // pad_enable
          | (4 << 22)                                 // pad_interval: 32 DWORDs (=64 bf16)
          | (3 << 25);                                // pad_amount:   4 DWORDs (=8 bf16)
    g1[1] = (DK_ << 16);                              // tensor_dim0[15:0] = 64
    g1[2] = (S_ << 16);                               // dim0 hi=0 | tensor_dim1[15:0]=2048
    g1[3] = (DK_ << 16);                              // dim1 hi=0 | tile_dim0 = 64
    g1[4] = 64;                                       // tile_dim1 = 64 | tile_dim2 = 0
    g1[5] = DK_;                                      // tensor_dim0_stride[31:0] = 64
    g1[6] = (int)(((unsigned)(S_ * DK_) & 0xffffu) << 16); // stride0 hi | stride1[15:0]
    g1[7] = (int)((unsigned)(S_ * DK_) >> 16);        // tensor_dim1_stride[47:16]
    i32x4 z = {0, 0, 0, 0};
#if __clang_major__ >= 23
    i32x8 z8 = {0, 0, 0, 0, 0, 0, 0, 0};
    __builtin_amdgcn_tensor_load_to_lds(g0, g1, z, z, z8, 0);
#else
    __builtin_amdgcn_tensor_load_to_lds(g0, g1, z, z, 0);
#endif
}
#endif // HAVE_TDM

// ---------------------------------------------------------------------------
// fp32 -> bf16 conversion (vector x4)
// ---------------------------------------------------------------------------
__global__ __launch_bounds__(256)
void cvt_f32_bf16_kernel(const float* __restrict__ x, bf16* __restrict__ y, int n4) {
    int i = blockIdx.x * blockDim.x + threadIdx.x;
    if (i < n4) {
        float4 f = ((const float4*)x)[i];
        bf16* o = y + 4 * (size_t)i;
        o[0] = (bf16)f.x; o[1] = (bf16)f.y; o[2] = (bf16)f.z; o[3] = (bf16)f.w;
    }
}

// ---------------------------------------------------------------------------
// bf16 GEMM: C[M,N] = A[M,K] * W[K,N], fp32 accumulation via WMMA.
// WG = 256 threads = 8 waves. WG tile 128(M) x 128(N), K step 32.
// Wave (wm,wn): wm in 0..3 (32 rows), wn in 0..1 (64 cols) -> 2x4 accumulators,
// 8 WMMAs per barrier pair, A-fragments reused across 4 B tiles.
// MODE 1: store bf16 scattered into [B,H,S,DK] (QKV). MODE 2: fp32 row-major.
// ---------------------------------------------------------------------------
template <int MODE>
__global__ __launch_bounds__(256)
void gemm_bf16_kernel(const bf16* __restrict__ A, const bf16* __restrict__ W,
                      void* __restrict__ Out, int M, int N, int K) {
    __shared__ bf16 As[128][40];   // A tile, row-major, K padded 32->40
    __shared__ bf16 Bt[128][40];   // W tile transposed (N x K), padded

    const int tid  = threadIdx.x;
    const int lane = tid & 31, wave = tid >> 5;
    const int half = lane >> 4, l15 = lane & 15;
    const int wm = wave >> 1, wn = wave & 1;
    const int m0 = blockIdx.x * 128;
    const int n0 = blockIdx.y * 128;

    v8f acc[2][4] = {};

    const int ar = tid >> 1, ac = (tid & 1) * 16;   // A staging: 128 rows x 32
    const int bk = tid >> 3, bc = (tid & 7) * 16;   // W staging: 32 rows x 128

    for (int k0 = 0; k0 < K; k0 += 32) {
        const bf16* ga = A + (size_t)(m0 + ar) * K + k0 + ac;
        const bf16* gw = W + (size_t)(k0 + bk) * N + n0 + bc;
        if (k0 + 32 < K) {        // prefetch next K-slice (global_prefetch_b8)
            __builtin_prefetch(ga + 32, 0, 0);
            __builtin_prefetch(gw + (size_t)32 * N, 0, 0);
        }
        // stage A tile: two 16B loads, contiguous LDS store
        *(v8bf*)&As[ar][ac]     = *(const v8bf*)ga;
        *(v8bf*)&As[ar][ac + 8] = *(const v8bf*)(ga + 8);
        // stage W tile transposed: coalesced global read, scatter into Bt[n][k]
        v8bf w0 = *(const v8bf*)gw;
        v8bf w1 = *(const v8bf*)(gw + 8);
#pragma unroll
        for (int i = 0; i < 8; ++i) {
            Bt[bc + i][bk]     = w0[i];
            Bt[bc + 8 + i][bk] = w1[i];
        }
        __syncthreads();

        v16bf af[2], bfrag[4];
#pragma unroll
        for (int i = 0; i < 2; ++i) {
            const bf16* p = &As[wm * 32 + i * 16 + l15][half * 8];
            af[i] = frag16(p, p + 16);
        }
#pragma unroll
        for (int j = 0; j < 4; ++j) {
            const bf16* p = &Bt[wn * 64 + j * 16 + l15][half * 8];
            bfrag[j] = frag16(p, p + 16);
        }
#pragma unroll
        for (int i = 0; i < 2; ++i)
#pragma unroll
            for (int j = 0; j < 4; ++j)
                acc[i][j] = wmma_bf16(af[i], bfrag[j], acc[i][j]);
        __syncthreads();
    }

    // store: C VGPR r -> row half*8+r, lane l15 -> column
#pragma unroll
    for (int i = 0; i < 2; ++i)
#pragma unroll
        for (int j = 0; j < 4; ++j)
#pragma unroll
            for (int r = 0; r < 8; ++r) {
                int row = m0 + wm * 32 + i * 16 + half * 8 + r;
                int col = n0 + wn * 64 + j * 16 + l15;
                if (MODE == 2) {
                    ((float*)Out)[(size_t)row * N + col] = acc[i][j][r];
                } else {
                    // row = b*S + s ; col = h*DK + dk  ->  [B,H,S,DK]
                    int b = row >> 11, s = row & (S_ - 1);
                    int h = col >> 6,  dk = col & (DK_ - 1);
                    ((bf16*)Out)[(((size_t)b * H_ + h) * S_ + s) * DK_ + dk] =
                        (bf16)acc[i][j][r];
                }
            }
}

// ---------------------------------------------------------------------------
// Flash attention: grid (S/64, B*H), 128 threads = 4 waves.
// Each wave owns 16 query rows; WG streams 64-key blocks of K/V.
// Q/K tiles staged by the Tensor Data Mover (wave 0 issues, TENSORcnt sync);
// V staged+transposed on the vector path (TDM cannot transpose).
// ---------------------------------------------------------------------------
__global__ __launch_bounds__(128)
void flash_attn_kernel(const bf16* __restrict__ Q, const bf16* __restrict__ Kb,
                       const bf16* __restrict__ Vb, const int* __restrict__ mask,
                       bf16* __restrict__ Ob) {
    __shared__ bf16 Qs[64][72];       // [q][dk]    (A layout for scores)
    __shared__ bf16 Ks[64][72];       // [key][dk]  (== transposed B for QK^T)
    __shared__ bf16 Vts[64][72];      // [dk][key]  (transposed B for P@V)
    __shared__ bf16 Ps[4][16][72];    // per-wave P tile [q][key]

    const int tid  = threadIdx.x;
    const int lane = tid & 31, wave = tid >> 5;
    const int half = lane >> 4, l15 = lane & 15;
    const int bh = blockIdx.y;                 // 0..63
    const int b = bh >> 4, h = bh & 15;
    const int q0 = blockIdx.x * 64;
    const size_t headBase = (size_t)bh * S_ * DK_;

    // stage Q block [64 x 64]
#if HAVE_TDM
    if (wave == 0)
        tdm_load_tile64(Q + headBase + (size_t)q0 * DK_,
                        (unsigned)(uintptr_t)&Qs[0][0]);
#else
    for (int v = tid; v < 64 * 8; v += 128) {
        int r = v >> 3, c = (v & 7) * 8;
        *(v8bf*)&Qs[r][c] = *(const v8bf*)(Q + headBase + (size_t)(q0 + r) * DK_ + c);
    }
#endif

    float mrun[8], lrun[8];
    v8f oacc[4] = {};
#pragma unroll
    for (int r = 0; r < 8; ++r) { mrun[r] = -1.0e30f; lrun[r] = 0.0f; }

    for (int j0 = 0; j0 < S_; j0 += 64) {
        // stage K block via TDM (wave 0); everyone stages V transposed
#if HAVE_TDM
        if (wave == 0)
            tdm_load_tile64(Kb + headBase + (size_t)j0 * DK_,
                            (unsigned)(uintptr_t)&Ks[0][0]);
#endif
        for (int v = tid; v < 64 * 8; v += 128) {
            int r = v >> 3, c = (v & 7) * 8;
#if !HAVE_TDM
            *(v8bf*)&Ks[r][c] =
                *(const v8bf*)(Kb + headBase + (size_t)(j0 + r) * DK_ + c);
#endif
            v8bf vv = *(const v8bf*)(Vb + headBase + (size_t)(j0 + r) * DK_ + c);
#pragma unroll
            for (int i = 0; i < 8; ++i) Vts[c + i][r] = vv[i];
        }
#if HAVE_TDM
        if (wave == 0) __builtin_amdgcn_s_wait_tensorcnt(0);   // Q (iter 0) + K done
#endif
        __syncthreads();

        // Q fragments for the two 32-wide dk chunks
        v16bf qa[2];
#pragma unroll
        for (int kc = 0; kc < 2; ++kc) {
            const bf16* p = &Qs[wave * 16 + l15][kc * 32 + half * 8];
            qa[kc] = frag16(p, p + 16);
        }

        // S = Q K^T : 4 key tiles x 2 dk chunks
        v8f sacc[4] = {};
#pragma unroll
        for (int t = 0; t < 4; ++t)
#pragma unroll
            for (int kc = 0; kc < 2; ++kc) {
                const bf16* p = &Ks[t * 16 + l15][kc * 32 + half * 8];
                sacc[t] = wmma_bf16(qa[kc], frag16(p, p + 16), sacc[t]);
            }

        // scale 1/sqrt(64) and key-axis mask bias
#pragma unroll
        for (int t = 0; t < 4; ++t) {
            int key = j0 + t * 16 + l15;
            float mb = mask[b * S_ + key] ? 0.0f : -1.0e9f;
#pragma unroll
            for (int r = 0; r < 8; ++r) sacc[t][r] = sacc[t][r] * 0.125f + mb;
        }

        // online softmax: per row (half*8+r), reduce across the 16 column lanes
#pragma unroll
        for (int r = 0; r < 8; ++r) {
            float mx = fmaxf(fmaxf(sacc[0][r], sacc[1][r]),
                             fmaxf(sacc[2][r], sacc[3][r]));
#pragma unroll
            for (int off = 8; off >= 1; off >>= 1)
                mx = fmaxf(mx, __shfl_xor(mx, off, 32));
            float mnew = fmaxf(mrun[r], mx);
            float corr = __expf(mrun[r] - mnew);
            mrun[r] = mnew;
            float ps = 0.0f;
#pragma unroll
            for (int t = 0; t < 4; ++t) {
                float p = __expf(sacc[t][r] - mnew);
                sacc[t][r] = p;
                ps += p;
            }
#pragma unroll
            for (int off = 8; off >= 1; off >>= 1) ps += __shfl_xor(ps, off, 32);
            lrun[r] = lrun[r] * corr + ps;
#pragma unroll
            for (int t = 0; t < 4; ++t) oacc[t][r] *= corr;
        }

        // transpose P (C layout) through LDS into A layout
#pragma unroll
        for (int t = 0; t < 4; ++t)
#pragma unroll
            for (int r = 0; r < 8; ++r)
                Ps[wave][half * 8 + r][t * 16 + l15] = (bf16)sacc[t][r];
        __syncthreads();

        // O += P @ V : 2 key chunks x 4 dk tiles
#pragma unroll
        for (int kc = 0; kc < 2; ++kc) {
            const bf16* pp = &Ps[wave][l15][kc * 32 + half * 8];
            v16bf pa = frag16(pp, pp + 16);
#pragma unroll
            for (int t = 0; t < 4; ++t) {
                const bf16* p = &Vts[t * 16 + l15][kc * 32 + half * 8];
                oacc[t] = wmma_bf16(pa, frag16(p, p + 16), oacc[t]);
            }
        }
        __syncthreads();
    }

    // epilogue: normalize and store as bf16 into [B,S,D] for the Wo GEMM
#pragma unroll
    for (int t = 0; t < 4; ++t)
#pragma unroll
        for (int r = 0; r < 8; ++r) {
            int q = q0 + wave * 16 + half * 8 + r;
            int d = h * DK_ + t * 16 + l15;
            float val = oacc[t][r] / lrun[r];
            Ob[((size_t)b * S_ + q) * D_ + d] = (bf16)val;
        }
}

// ---------------------------------------------------------------------------
// Host-side orchestration
// ---------------------------------------------------------------------------
extern "C" void kernel_launch(void* const* d_in, const int* in_sizes, int n_in,
                              void* d_out, int out_size, void* d_ws, size_t ws_size,
                              hipStream_t stream) {
    const float* X    = (const float*)d_in[0];   // [B,S,D]
    const int*   mask = (const int*)  d_in[1];   // [B,S]
    const float* Wq   = (const float*)d_in[2];
    const float* Wk   = (const float*)d_in[3];
    const float* Wv   = (const float*)d_in[4];
    const float* Wo   = (const float*)d_in[5];
    float* out = (float*)d_out;

    const size_t M  = (size_t)B_ * S_;       // 8192
    const size_t XE = M * D_;                // 8M elems
    const size_t WE = (size_t)D_ * D_;       // 1M elems

    // workspace carve (bf16): X, 4 weights, Q, K, V, attn-out  => ~88 MB
    char* ws = (char*)d_ws;
    bf16* Xb  = (bf16*)ws;  ws += XE * sizeof(bf16);
    bf16* Wqb = (bf16*)ws;  ws += WE * sizeof(bf16);
    bf16* Wkb = (bf16*)ws;  ws += WE * sizeof(bf16);
    bf16* Wvb = (bf16*)ws;  ws += WE * sizeof(bf16);
    bf16* Wob = (bf16*)ws;  ws += WE * sizeof(bf16);
    bf16* Qb  = (bf16*)ws;  ws += XE * sizeof(bf16);
    bf16* Kb  = (bf16*)ws;  ws += XE * sizeof(bf16);
    bf16* Vb  = (bf16*)ws;  ws += XE * sizeof(bf16);
    bf16* Ab  = (bf16*)ws;  ws += XE * sizeof(bf16);

    // fp32 -> bf16
    {
        int n4 = (int)(XE / 4);
        cvt_f32_bf16_kernel<<<(n4 + 255) / 256, 256, 0, stream>>>(X, Xb, n4);
        int w4 = (int)(WE / 4);
        cvt_f32_bf16_kernel<<<(w4 + 255) / 256, 256, 0, stream>>>(Wq, Wqb, w4);
        cvt_f32_bf16_kernel<<<(w4 + 255) / 256, 256, 0, stream>>>(Wk, Wkb, w4);
        cvt_f32_bf16_kernel<<<(w4 + 255) / 256, 256, 0, stream>>>(Wv, Wvb, w4);
        cvt_f32_bf16_kernel<<<(w4 + 255) / 256, 256, 0, stream>>>(Wo, Wob, w4);
    }

    // Q/K/V projections -> [B,H,S,DK] bf16
    dim3 ggrid((unsigned)(M / 128), (unsigned)(D_ / 128));
    gemm_bf16_kernel<1><<<ggrid, 256, 0, stream>>>(Xb, Wqb, (void*)Qb,
                                                   (int)M, D_, D_);
    gemm_bf16_kernel<1><<<ggrid, 256, 0, stream>>>(Xb, Wkb, (void*)Kb,
                                                   (int)M, D_, D_);
    gemm_bf16_kernel<1><<<ggrid, 256, 0, stream>>>(Xb, Wvb, (void*)Vb,
                                                   (int)M, D_, D_);

    // flash attention -> [B,S,D] bf16
    dim3 agrid((unsigned)(S_ / 64), (unsigned)(B_ * H_));
    flash_attn_kernel<<<agrid, 128, 0, stream>>>(Qb, Kb, Vb, mask, Ab);

    // output projection -> fp32 [B,S,D]
    gemm_bf16_kernel<2><<<ggrid, 256, 0, stream>>>(Ab, Wob, (void*)out,
                                                   (int)M, D_, D_);
}
